// Block_50113678410401
// MI455X (gfx1250) — compile-verified, matrix operand order checked by reading, not run on hardware
//
#include <hip/hip_runtime.h>

typedef __attribute__((ext_vector_type(16))) _Float16 v16h;
typedef __attribute__((ext_vector_type(8)))  float    v8f;
typedef __attribute__((ext_vector_type(4)))  unsigned int u32x4;
typedef __attribute__((ext_vector_type(8)))  int      i32x8;
typedef __attribute__((ext_vector_type(4)))  int      i32x4;

constexpr int B_ = 8;
constexpr int T_ = 1024;
constexpr int C_ = 1536;
constexpr int H_ = 12;
constexpr int D_ = 128;
constexpr float EPS_ = 1e-5f;

#if defined(__has_builtin)
#if __has_builtin(__builtin_amdgcn_tensor_load_to_lds) && \
    __has_builtin(__builtin_amdgcn_s_wait_tensorcnt)
#define USE_TDM 1
#endif
#endif

#ifdef USE_TDM
// ---------------------------------------------------------------------------
// TDM 2D tile load: global (row-major, stride0 elems between rows) -> LDS,
// f16 elements, optional LDS row padding (pad codes per CDNA5 D# spec:
// pad_interval: 0=2,1=4,2=8,3=16,4=32,5=64 DWORDs; pad_amount: code+1 DWORDs).
// Descriptor per CDNA5 ISA. Issued once per calling wave
// (EXEC ignored); completion via TENSORcnt.
// ---------------------------------------------------------------------------
__device__ inline void tdm_load_2d(void* lds_ptr, const void* gptr,
                                   unsigned tile_d0, unsigned tile_d1,
                                   unsigned long long stride0_elems,
                                   unsigned pad_interval_code,
                                   unsigned pad_amount_code) {
  unsigned lds_off = (unsigned)(size_t)lds_ptr;  // addr[31:0] == LDS offset
  unsigned long long ga = (unsigned long long)(size_t)gptr;
  u32x4 g0;
  g0[0] = 1u;                                    // count=1, user descriptor
  g0[1] = lds_off;                               // lds_addr [63:32]
  g0[2] = (unsigned)(ga & 0xffffffffu);          // global_addr lo
  g0[3] = (unsigned)((ga >> 32) & 0x01ffffffu) | (2u << 30);  // hi | type=2
  unsigned w0 = (1u << 16)                       // data_size = 2 bytes
              | (1u << 20)                       // pad_enable
              | (pad_interval_code << 22)
              | (pad_amount_code << 25);
  i32x8 g1;
  g1[0] = (int)w0;
  // tensor_dim0 = tile_d0 (bits 79:48), tensor_dim1 = tile_d1 (bits 111:80)
  g1[1] = (int)((tile_d0 & 0xffffu) << 16);
  g1[2] = (int)(((tile_d0 >> 16) & 0xffffu) | ((tile_d1 & 0xffffu) << 16));
  g1[3] = (int)(((tile_d1 >> 16) & 0xffffu) | ((tile_d0 & 0xffffu) << 16)); // tile_dim0
  g1[4] = (int)(tile_d1 & 0xffffu);              // tile_dim1 (tile_dim2 = 0)
  g1[5] = (int)(unsigned)(stride0_elems & 0xffffffffull);      // dim0 stride lo
  g1[6] = (int)(unsigned)((stride0_elems >> 32) & 0xffffull);  // stride hi
  g1[7] = 0;                                                   // dim1 stride = 0
  i32x4 z4 = {0, 0, 0, 0};
#if __clang_major__ >= 23
  i32x8 z8 = {0, 0, 0, 0, 0, 0, 0, 0};
  __builtin_amdgcn_tensor_load_to_lds(g0, g1, z4, z4, z8, 0);
#else
  __builtin_amdgcn_tensor_load_to_lds(g0, g1, z4, z4, 0);
#endif
}
#endif  // USE_TDM

// ---------------------------------------------------------------------------
// Fragment load helper: 16-bit A/B fragment (16x32) from LDS tile stored as
// (rows x K) row-major with row stride `ld` (multiple of 8 halfs).
// Lane L: row = rowbase + (L&15); halves: K = kk + 8*(L>>4) .. +7 and +16.
// ---------------------------------------------------------------------------
__device__ inline v16h ldsfrag(const _Float16* t, int ld, int rowbase, int kk,
                               int half, int l15) {
  const _Float16* p = t + (rowbase + l15) * ld + kk + half * 8;
  union { v16h h; float4 f[2]; } u;
  u.f[0] = *(const float4*)(p);
  u.f[1] = *(const float4*)(p + 16);
  return u.h;
}

__device__ inline void zero_acc(v8f a[2][4]) {
#pragma unroll
  for (int i = 0; i < 2; ++i)
#pragma unroll
    for (int j = 0; j < 4; ++j)
#pragma unroll
      for (int e = 0; e < 8; ++e) a[i][j][e] = 0.0f;
}

// ---------------------------------------------------------------------------
// Weight conversion kernels (run every call; trivial bandwidth)
// ---------------------------------------------------------------------------
__global__ __launch_bounds__(256) void k_pack_qkv(
    const float* __restrict__ wq, const float* __restrict__ wk,
    const float* __restrict__ wv, _Float16* __restrict__ dst) {
  int idx = blockIdx.x * 256 + threadIdx.x;
  const int per = H_ * C_ * D_;
  if (idx >= 3 * per) return;
  int sel = idx / per;
  int r = idx % per;
  int h = r / (C_ * D_);
  int r2 = r % (C_ * D_);
  int c = r2 / D_;
  int d = r2 % D_;
  const float* w = (sel == 0) ? wq : (sel == 1) ? wk : wv;
  float v = w[(size_t)h * C_ * D_ + (size_t)c * D_ + d];
  int n = sel * C_ + h * D_ + d;          // output column index
  dst[(size_t)n * C_ + c] = (_Float16)v;  // stored transposed: (N, K)
}

__global__ __launch_bounds__(256) void k_transpose(
    const float* __restrict__ src, _Float16* __restrict__ dst, int K, int N) {
  int idx = blockIdx.x * 256 + threadIdx.x;
  if (idx >= K * N) return;
  int k = idx / N, n = idx % N;
  dst[(size_t)n * K + k] = (_Float16)src[idx];
}

// ---------------------------------------------------------------------------
// LayerNorm per row (C=1536), output f16 for GEMM A input
// ---------------------------------------------------------------------------
__global__ __launch_bounds__(256) void k_ln(
    const float* __restrict__ x, const float* __restrict__ g,
    const float* __restrict__ bt, _Float16* __restrict__ out) {
  __shared__ float red[256];
  __shared__ float s_m, s_inv;
  size_t row = blockIdx.x;
  const float* xr = x + row * C_;
  int tid = threadIdx.x;
  float sum = 0.f, sq = 0.f;
  for (int c = tid; c < C_; c += 256) {
    float v = xr[c];
    sum += v; sq += v * v;
  }
  red[tid] = sum; __syncthreads();
  for (int s = 128; s > 0; s >>= 1) { if (tid < s) red[tid] += red[tid + s]; __syncthreads(); }
  float tot = red[0]; __syncthreads();
  red[tid] = sq; __syncthreads();
  for (int s = 128; s > 0; s >>= 1) { if (tid < s) red[tid] += red[tid + s]; __syncthreads(); }
  if (tid == 0) {
    float m = tot / C_;
    float var = red[0] / C_ - m * m;
    s_m = m; s_inv = rsqrtf(var + EPS_);
  }
  __syncthreads();
  float m = s_m, inv = s_inv;
  for (int c = tid; c < C_; c += 256)
    out[row * C_ + c] = (_Float16)((xr[c] - m) * inv * g[c] + bt[c]);
}

// ---------------------------------------------------------------------------
// BatchNorm1d over (B, C) per t-channel, biased variance, in place
// ---------------------------------------------------------------------------
__global__ __launch_bounds__(256) void k_bn(
    float* __restrict__ x, const float* __restrict__ g,
    const float* __restrict__ bt) {
  __shared__ float red[256];
  __shared__ float s_m, s_inv;
  int t = blockIdx.x, tid = threadIdx.x;
  const int NE = B_ * C_;
  float sum = 0.f, sq = 0.f;
  for (int i = tid; i < NE; i += 256) {
    int b = i / C_, c = i % C_;
    float v = x[((size_t)(b * T_ + t)) * C_ + c];
    sum += v; sq += v * v;
  }
  red[tid] = sum; __syncthreads();
  for (int s = 128; s > 0; s >>= 1) { if (tid < s) red[tid] += red[tid + s]; __syncthreads(); }
  float tot = red[0]; __syncthreads();
  red[tid] = sq; __syncthreads();
  for (int s = 128; s > 0; s >>= 1) { if (tid < s) red[tid] += red[tid + s]; __syncthreads(); }
  if (tid == 0) {
    float m = tot / NE;
    float var = red[0] / NE - m * m;
    s_m = m; s_inv = rsqrtf(var + EPS_);
  }
  __syncthreads();
  float m = s_m, inv = s_inv, gg = g[t], bb = bt[t];
  for (int i = tid; i < NE; i += 256) {
    int b = i / C_, c = i % C_;
    size_t idx = ((size_t)(b * T_ + t)) * C_ + c;
    x[idx] = (x[idx] - m) * inv * gg + bb;
  }
}

// ---------------------------------------------------------------------------
// Generic WMMA GEMM: out = A(MxK,f16,row-major) * Bt(NxK,f16,row-major)^T
// Block tile 128x128, BK=64, 8 waves -> each wave 32x64 (2x4 wmma frags).
// TDM path: double-buffered LDS tiles, wave 0 drives the DMA (TENSORcnt),
// overlapping the next tile's DMA with WMMA on the current one.
// EPI 0: scatter q/k/vT   EPI 1: +bias +resid -> f32   EPI 2: relu(+bias)->f16
// ---------------------------------------------------------------------------
template <int EPI>
__global__ __launch_bounds__(256) void k_gemm(
    const _Float16* __restrict__ A, const _Float16* __restrict__ Bt,
    int M, int N, int K, const float* __restrict__ bias,
    const float* __restrict__ resid, void* out0, void* out1, void* out2) {
  constexpr int LDT = 72;  // padded K stride (multiple of 8)
  const int tid = threadIdx.x;
  const int lane = tid & 31, wid = tid >> 5;
  const int wm = (wid >> 1) * 32, wn = (wid & 1) * 64;
  const int half = lane >> 4, l15 = lane & 15;
  const int m0 = blockIdx.y * 128, n0 = blockIdx.x * 128;

  v8f acc[2][4];
  zero_acc(acc);

#ifdef USE_TDM
  __shared__ _Float16 as[2][128 * LDT];
  __shared__ _Float16 bs[2][128 * LDT];
  // pad codes: 32 DWORDs (64 halfs) data, 4 DWORDs (8 halfs) pad -> stride 72
  if (wid == 0) {
    tdm_load_2d(&as[0][0], &A[(size_t)m0 * K], 64, 128, (unsigned)K, 4, 3);
    tdm_load_2d(&bs[0][0], &Bt[(size_t)n0 * K], 64, 128, (unsigned)K, 4, 3);
  }
  int cur = 0;
  for (int k0 = 0; k0 < K; k0 += 64) {
    if (wid == 0) __builtin_amdgcn_s_wait_tensorcnt(0);
    __syncthreads();  // tile `cur` ready; previous compute on `cur^1` done
    if (wid == 0 && (k0 + 64) < K) {
      tdm_load_2d(&as[cur ^ 1][0], &A[(size_t)m0 * K + k0 + 64], 64, 128,
                  (unsigned)K, 4, 3);
      tdm_load_2d(&bs[cur ^ 1][0], &Bt[(size_t)n0 * K + k0 + 64], 64, 128,
                  (unsigned)K, 4, 3);
    }
#pragma unroll
    for (int kk = 0; kk < 64; kk += 32) {
      v16h af[2], bf[4];
#pragma unroll
      for (int mi = 0; mi < 2; ++mi)
        af[mi] = ldsfrag(&as[cur][0], LDT, wm + mi * 16, kk, half, l15);
#pragma unroll
      for (int ni = 0; ni < 4; ++ni)
        bf[ni] = ldsfrag(&bs[cur][0], LDT, wn + ni * 16, kk, half, l15);
#pragma unroll
      for (int mi = 0; mi < 2; ++mi)
#pragma unroll
        for (int ni = 0; ni < 4; ++ni)
          acc[mi][ni] = __builtin_amdgcn_wmma_f32_16x16x32_f16(
              false, af[mi], false, bf[ni], (short)0, acc[mi][ni], false, false);
    }
    cur ^= 1;
  }
#else
  __shared__ _Float16 as[128 * LDT];
  __shared__ _Float16 bs[128 * LDT];
  for (int k0 = 0; k0 < K; k0 += 64) {
#pragma unroll
    for (int c = 0; c < 4; ++c) {
      int idx = tid + c * 256;  // 1024 chunks of 8 halfs
      int row = idx >> 3, col = (idx & 7) * 8;
      *(float4*)(&as[row * LDT + col]) =
          *(const float4*)(&A[(size_t)(m0 + row) * K + k0 + col]);
      *(float4*)(&bs[row * LDT + col]) =
          *(const float4*)(&Bt[(size_t)(n0 + row) * K + k0 + col]);
    }
    __syncthreads();
#pragma unroll
    for (int kk = 0; kk < 64; kk += 32) {
      v16h af[2], bf[4];
#pragma unroll
      for (int mi = 0; mi < 2; ++mi) af[mi] = ldsfrag(as, LDT, wm + mi * 16, kk, half, l15);
#pragma unroll
      for (int ni = 0; ni < 4; ++ni) bf[ni] = ldsfrag(bs, LDT, wn + ni * 16, kk, half, l15);
#pragma unroll
      for (int mi = 0; mi < 2; ++mi)
#pragma unroll
        for (int ni = 0; ni < 4; ++ni)
          acc[mi][ni] = __builtin_amdgcn_wmma_f32_16x16x32_f16(
              false, af[mi], false, bf[ni], (short)0, acc[mi][ni], false, false);
    }
    __syncthreads();
  }
#endif

#pragma unroll
  for (int mi = 0; mi < 2; ++mi)
#pragma unroll
    for (int ni = 0; ni < 4; ++ni)
#pragma unroll
      for (int j = 0; j < 8; ++j) {
        int row = m0 + wm + mi * 16 + j + 8 * half;
        int col = n0 + wn + ni * 16 + l15;
        float v = acc[mi][ni][j];
        if constexpr (EPI == 0) {  // q/k/vT scatter
          int b = row >> 10, t = row & 1023;
          int sel = col / C_, rem = col % C_;
          int hh = rem >> 7, d = rem & 127;
          _Float16 hv = (_Float16)v;
          if (sel == 0)
            ((_Float16*)out0)[((size_t)((b * H_ + hh) * T_ + t)) * D_ + d] = hv;
          else if (sel == 1)
            ((_Float16*)out1)[((size_t)((b * H_ + hh) * T_ + t)) * D_ + d] = hv;
          else
            ((_Float16*)out2)[((size_t)((b * H_ + hh) * D_ + d)) * T_ + t] = hv;
        } else if constexpr (EPI == 1) {  // +bias +resid -> f32
          size_t idx = (size_t)row * N + col;
          ((float*)out0)[idx] = v + bias[col] + resid[idx];
        } else {  // relu(+bias) -> f16
          float r = v + bias[col];
          ((_Float16*)out0)[(size_t)row * N + col] = (_Float16)(r > 0.f ? r : 0.f);
        }
      }
}

// ---------------------------------------------------------------------------
// Flash-style causal attention. One block per (b, h, 128-row q tile).
// q,k stored (B,H,T,D) f16; v stored transposed (B,H,D,T) f16.
// Output o written as (B,T,C) f16 with C index = h*128 + d.
// k/v tiles staged via TDM when available.
// ---------------------------------------------------------------------------
__global__ __launch_bounds__(256) void k_attn(
    const _Float16* __restrict__ q, const _Float16* __restrict__ k,
    const _Float16* __restrict__ vt, _Float16* __restrict__ o) {
  constexpr int LD = 136;    // f16 tile K-stride (64+4 DWORD pad per row)
  constexpr int LDSS = 132;  // f32 score stride
  __shared__ _Float16 qs[128 * LD];
  __shared__ _Float16 ks[128 * LD];
  __shared__ _Float16 vs[128 * LD];
  __shared__ _Float16 ps[128 * LD];
  __shared__ float Sb[128 * LDSS];
  __shared__ float mrun[128], lrun[128], scl[128];

  const int tid = threadIdx.x, lane = tid & 31, wid = tid >> 5;
  const int wm = (wid >> 1) * 32, wn = (wid & 1) * 64;
  const int half = lane >> 4, l15 = lane & 15;
  const int qi = blockIdx.x, hh = blockIdx.y, b = blockIdx.z;

  const _Float16* qbase = q + ((size_t)(b * H_ + hh) * T_ + qi * 128) * D_;
  const _Float16* kbase = k + (size_t)(b * H_ + hh) * T_ * D_;
  const _Float16* vbase = vt + (size_t)(b * H_ + hh) * D_ * T_;

#ifdef USE_TDM
  if (wid == 0) tdm_load_2d(qs, qbase, 128, 128, (unsigned)D_, 5, 3);
#else
#pragma unroll
  for (int c = 0; c < 8; ++c) {  // 128x128 f16 q tile
    int idx = tid + c * 256;
    int row = idx >> 4, col = (idx & 15) * 8;
    *(float4*)(&qs[row * LD + col]) = *(const float4*)(&qbase[(size_t)row * D_ + col]);
  }
#endif
  if (tid < 128) { mrun[tid] = -3.0e38f; lrun[tid] = 0.f; }
  v8f acc[2][4];
  zero_acc(acc);
#ifndef USE_TDM
  __syncthreads();
#endif

  const float sscale = 0.08838834764831845f;  // 1/sqrt(128)

  for (int sj = 0; sj <= qi; ++sj) {
#ifdef USE_TDM
    if (wid == 0) {
      tdm_load_2d(ks, &kbase[(size_t)sj * 128 * D_], 128, 128, (unsigned)D_, 5, 3);
      tdm_load_2d(vs, &vbase[sj * 128], 128, 128, (unsigned)T_, 5, 3);
      __builtin_amdgcn_s_wait_tensorcnt(0);  // also covers the q tile (sj==0)
    }
    __syncthreads();
#else
#pragma unroll
    for (int c = 0; c < 8; ++c) {
      int idx = tid + c * 256;
      int row = idx >> 4, col = (idx & 15) * 8;
      *(float4*)(&ks[row * LD + col]) =
          *(const float4*)(&kbase[(size_t)(sj * 128 + row) * D_ + col]);
      *(float4*)(&vs[row * LD + col]) =
          *(const float4*)(&vbase[(size_t)row * T_ + sj * 128 + col]);
    }
    __syncthreads();
#endif

    // S = q * k^T  (A = q tile, Bt = k tile: rows are s-index, cols are D)
    v8f sf[2][4];
    zero_acc(sf);
#pragma unroll
    for (int kk = 0; kk < 128; kk += 32) {
      v16h af[2], bf[4];
#pragma unroll
      for (int mi = 0; mi < 2; ++mi) af[mi] = ldsfrag(qs, LD, wm + mi * 16, kk, half, l15);
#pragma unroll
      for (int ni = 0; ni < 4; ++ni) bf[ni] = ldsfrag(ks, LD, wn + ni * 16, kk, half, l15);
#pragma unroll
      for (int mi = 0; mi < 2; ++mi)
#pragma unroll
        for (int ni = 0; ni < 4; ++ni)
          sf[mi][ni] = __builtin_amdgcn_wmma_f32_16x16x32_f16(
              false, af[mi], false, bf[ni], (short)0, sf[mi][ni], false, false);
    }
#pragma unroll
    for (int mi = 0; mi < 2; ++mi)
#pragma unroll
      for (int ni = 0; ni < 4; ++ni)
#pragma unroll
        for (int j = 0; j < 8; ++j) {
          int r = wm + mi * 16 + j + 8 * half;
          int c2 = wn + ni * 16 + l15;
          Sb[r * LDSS + c2] = sf[mi][ni][j] * sscale;
        }
    __syncthreads();

    // Online softmax update, one thread per row
    if (tid < 128) {
      int r = tid;
      int cnt = (sj == qi) ? (r + 1) : 128;  // causal: s <= t
      float mx = -3.0e38f;
      for (int c2 = 0; c2 < cnt; ++c2) mx = fmaxf(mx, Sb[r * LDSS + c2]);
      float mnew = fmaxf(mrun[r], mx);
      float psum = 0.f;
      for (int c2 = 0; c2 < 128; ++c2) {
        float p = (c2 < cnt) ? __expf(Sb[r * LDSS + c2] - mnew) : 0.f;
        ps[r * LD + c2] = (_Float16)p;
        psum += p;
      }
      float sc = __expf(mrun[r] - mnew);
      lrun[r] = lrun[r] * sc + psum;
      scl[r] = sc;
      mrun[r] = mnew;
    }
    __syncthreads();

    // rescale accumulator, then acc += P * V  (Bt = vT tile: rows d, cols s)
#pragma unroll
    for (int mi = 0; mi < 2; ++mi)
#pragma unroll
      for (int ni = 0; ni < 4; ++ni)
#pragma unroll
        for (int j = 0; j < 8; ++j)
          acc[mi][ni][j] *= scl[wm + mi * 16 + j + 8 * half];
#pragma unroll
    for (int kk = 0; kk < 128; kk += 32) {
      v16h af[2], bf[4];
#pragma unroll
      for (int mi = 0; mi < 2; ++mi) af[mi] = ldsfrag(ps, LD, wm + mi * 16, kk, half, l15);
#pragma unroll
      for (int ni = 0; ni < 4; ++ni) bf[ni] = ldsfrag(vs, LD, wn + ni * 16, kk, half, l15);
#pragma unroll
      for (int mi = 0; mi < 2; ++mi)
#pragma unroll
        for (int ni = 0; ni < 4; ++ni)
          acc[mi][ni] = __builtin_amdgcn_wmma_f32_16x16x32_f16(
              false, af[mi], false, bf[ni], (short)0, acc[mi][ni], false, false);
    }
    __syncthreads();  // protects ks/vs (and ps) before next iteration's DMA
  }

  // finalize: divide by l, write o (B,T,C) f16
#pragma unroll
  for (int mi = 0; mi < 2; ++mi)
#pragma unroll
    for (int ni = 0; ni < 4; ++ni)
#pragma unroll
      for (int j = 0; j < 8; ++j) {
        int r = wm + mi * 16 + j + 8 * half;
        int c2 = wn + ni * 16 + l15;
        float v = acc[mi][ni][j] / lrun[r];
        int t = qi * 128 + r;
        o[((size_t)(b * T_ + t)) * C_ + hh * D_ + c2] = (_Float16)v;
      }
}

// ---------------------------------------------------------------------------
extern "C" void kernel_launch(void* const* d_in, const int* in_sizes, int n_in,
                              void* d_out, int out_size, void* d_ws,
                              size_t ws_size, hipStream_t stream) {
  (void)in_sizes; (void)n_in; (void)out_size; (void)ws_size;
  const float* x    = (const float*)d_in[0];
  const float* wq   = (const float*)d_in[1];
  const float* wk   = (const float*)d_in[2];
  const float* wv   = (const float*)d_in[3];
  const float* wo   = (const float*)d_in[4];
  const float* bo   = (const float*)d_in[5];
  const float* ln1g = (const float*)d_in[6];
  const float* ln1b = (const float*)d_in[7];
  const float* ln2g = (const float*)d_in[8];
  const float* ln2b = (const float*)d_in[9];
  const float* w1   = (const float*)d_in[10];
  const float* b1   = (const float*)d_in[11];
  const float* w2   = (const float*)d_in[12];
  const float* b2   = (const float*)d_in[13];
  const float* bn1g = (const float*)d_in[14];
  const float* bn1b = (const float*)d_in[15];
  const float* bn2g = (const float*)d_in[16];
  const float* bn2b = (const float*)d_in[17];
  float* out = (float*)d_out;

  const size_t M = (size_t)B_ * T_;  // 8192
  const size_t S_h     = M * C_ * 2;                 // 25.2 MB (h, reused as h2)
  const size_t S_wqkvT = (size_t)3 * C_ * C_ * 2;    // 14.2 MB
  const size_t S_qkv   = (size_t)B_ * H_ * T_ * D_ * 2;  // 25.2 MB each
  const size_t S_woT   = (size_t)C_ * C_ * 2;        // 4.7 MB
  const size_t S_x1    = M * C_ * 4;                 // 50.3 MB
  const size_t S_w1T   = (size_t)C_ * 4 * C_ * 2;    // 18.9 MB

  char* ws = (char*)d_ws;
  _Float16* hbuf  = (_Float16*)(ws);
  _Float16* wqkvT = (_Float16*)(ws + S_h);
  _Float16* qbuf  = (_Float16*)(ws + S_h + S_wqkvT);
  _Float16* kbuf  = (_Float16*)(ws + S_h + S_wqkvT + S_qkv);
  _Float16* vtbuf = (_Float16*)(ws + S_h + S_wqkvT + 2 * S_qkv);
  _Float16* obuf  = (_Float16*)(ws + S_h + S_wqkvT + 3 * S_qkv);
  _Float16* ubuf  = qbuf;  // FFN hidden (8192x6144 f16) aliases q..o region
  _Float16* woT   = (_Float16*)(ws + S_h + S_wqkvT + 4 * S_qkv);
  float*    x1    = (float*)(ws + S_h + S_wqkvT + 4 * S_qkv + S_woT);
  _Float16* w1T   = (_Float16*)(ws + S_h + S_wqkvT + 4 * S_qkv + S_woT + S_x1);
  _Float16* w2T   = (_Float16*)(ws + S_h + S_wqkvT + 4 * S_qkv + S_woT + S_x1 + S_w1T);

  // weight conversion (f32 -> f16, transposed to (N,K))
  k_pack_qkv<<<(3 * H_ * C_ * D_ + 255) / 256, 256, 0, stream>>>(wq, wk, wv, wqkvT);
  k_transpose<<<(C_ * C_ + 255) / 256, 256, 0, stream>>>(wo, woT, C_, C_);
  k_transpose<<<(C_ * 4 * C_ + 255) / 256, 256, 0, stream>>>(w1, w1T, C_, 4 * C_);
  k_transpose<<<(4 * C_ * C_ + 255) / 256, 256, 0, stream>>>(w2, w2T, 4 * C_, C_);

  // h = LN1(x)
  k_ln<<<(int)M, 256, 0, stream>>>(x, ln1g, ln1b, hbuf);
  // q,k,vT = h @ Wqkv
  k_gemm<0><<<dim3(36, 64), 256, 0, stream>>>(hbuf, wqkvT, (int)M, 3 * C_, C_,
                                              nullptr, nullptr, qbuf, kbuf, vtbuf);
  // causal attention
  k_attn<<<dim3(T_ / 128, H_, B_), 256, 0, stream>>>(qbuf, kbuf, vtbuf, obuf);
  // x1 = x + o @ Wo + bo
  k_gemm<1><<<dim3(12, 64), 256, 0, stream>>>(obuf, woT, (int)M, C_, C_,
                                              bo, x, x1, nullptr, nullptr);
  // x1 = BN1(x1)
  k_bn<<<T_, 256, 0, stream>>>(x1, bn1g, bn1b);
  // h2 = LN2(x1)
  k_ln<<<(int)M, 256, 0, stream>>>(x1, ln2g, ln2b, hbuf);
  // u = relu(h2 @ W1 + b1)
  k_gemm<2><<<dim3(48, 64), 256, 0, stream>>>(hbuf, w1T, (int)M, 4 * C_, C_,
                                              b1, nullptr, ubuf, nullptr, nullptr);
  // out = x1 + u @ W2 + b2
  k_gemm<1><<<dim3(12, 64), 256, 0, stream>>>(ubuf, w2T, (int)M, C_, 4 * C_,
                                              b2, x1, out, nullptr, nullptr);
  // out = BN2(out)
  k_bn<<<T_, 256, 0, stream>>>(out, bn2g, bn2b);
}
